// GraphEncoder_11553462026276
// MI455X (gfx1250) — matrix operand order repaired and checked by reading, not compiled, not run on hardware
//
#include <hip/hip_runtime.h>
#include <hip/hip_bf16.h>

typedef __attribute__((ext_vector_type(16))) __bf16 bf16x16;
typedef __attribute__((ext_vector_type(8)))  __bf16 bf16x8;
typedef __attribute__((ext_vector_type(8)))  float  floatx8;

// ---------------------------------------------------------------------------
// Convert 16 fp32 (two contiguous chunks of 8) into a WMMA bf16 A/B fragment.
// A-fragment layout (16-bit, 16x32): lane<16 holds row M=lane, K = {k0+0..7} u
// {k0+16..23}; lane>=16 holds same row, K = {k0+8..15} u {k0+24..31}.  The two
// chunk base pointers are passed in already resolved per-lane.
// ---------------------------------------------------------------------------
__device__ __forceinline__ bf16x16 cvt_frag(const float* __restrict__ p1,
                                            const float* __restrict__ p2) {
  float4 a0 = *(const float4*)(p1);
  float4 a1 = *(const float4*)(p1 + 4);
  float4 a2 = *(const float4*)(p2);
  float4 a3 = *(const float4*)(p2 + 4);
  bf16x16 a;
  a[0]  = (__bf16)a0.x; a[1]  = (__bf16)a0.y; a[2]  = (__bf16)a0.z; a[3]  = (__bf16)a0.w;
  a[4]  = (__bf16)a1.x; a[5]  = (__bf16)a1.y; a[6]  = (__bf16)a1.z; a[7]  = (__bf16)a1.w;
  a[8]  = (__bf16)a2.x; a[9]  = (__bf16)a2.y; a[10] = (__bf16)a2.z; a[11] = (__bf16)a2.w;
  a[12] = (__bf16)a3.x; a[13] = (__bf16)a3.y; a[14] = (__bf16)a3.z; a[15] = (__bf16)a3.w;
  return a;
}

// ---------------------------------------------------------------------------
// Pack a row-major fp32 [K x N] weight into bf16 WMMA B-fragment layout:
// tile (kt, nt) occupies 512 elems; within a tile, lane l element i holds
// B[kt*32 + (i>>3)*16 + (l>>4)*8 + (i&7)][nt*16 + (l&15)].  Columns >= N are zero-padded.
// ---------------------------------------------------------------------------
__global__ void pack_b_kernel(const float* __restrict__ B, __bf16* __restrict__ Bp,
                              int K, int N, int ntTotal) {
  int t = blockIdx.x * blockDim.x + threadIdx.x;
  int total = (K >> 5) * ntTotal * 512;
  if (t >= total) return;
  int i    = t & 15;
  int lane = (t >> 4) & 31;
  int tile = t >> 9;
  int nt   = tile % ntTotal;
  int kt   = tile / ntTotal;
  int col  = nt * 16 + (lane & 15);
  int krow = kt * 32 + (i >> 3) * 16 + (lane >> 4) * 8 + (i & 7);
  float v = (col < N) ? B[(size_t)krow * N + col] : 0.0f;
  Bp[t] = (__bf16)v;
}

// ---------------------------------------------------------------------------
// Generic bf16 WMMA GEMM: C[M x N] (+bias, optional ReLU) = A[M x K] @ B.
// 4 waves/block, 16 rows/wave, NT 16-col tiles starting at ntBase.
// ---------------------------------------------------------------------------
template <int NT, bool RELU>
__global__ void __launch_bounds__(128)
gemm_bf16_wmma(const float* __restrict__ A, const __bf16* __restrict__ Bp,
               const float* __restrict__ bias, float* __restrict__ C,
               int M, int K, int N, int ntBase, int ntTotal) {
  const int lane = threadIdx.x & 31;
  const int wv   = threadIdx.x >> 5;
  const int half = lane >> 4;
  const int l16  = lane & 15;
  const int rowBase = (blockIdx.x * 4 + wv) * 16;
  if (rowBase >= M) return;  // wave-uniform: EXEC stays all-ones for WMMA

  floatx8 acc[NT];
#pragma unroll
  for (int nt = 0; nt < NT; ++nt) acc[nt] = 0.0f;

  const float* arow = A + (size_t)(rowBase + l16) * K;
  for (int k0 = 0; k0 < K; k0 += 32) {
    const bf16x16 a = cvt_frag(arow + k0 + half * 8, arow + k0 + 16 + half * 8);
    const __bf16* bt = Bp + ((size_t)(k0 >> 5) * ntTotal + ntBase) * 512 + lane * 16;
#pragma unroll
    for (int nt = 0; nt < NT; ++nt) {
      const bf16x16 b = *(const bf16x16*)(bt + (size_t)nt * 512);
      acc[nt] = __builtin_amdgcn_wmma_f32_16x16x32_bf16(false, a, false, b,
                                                        (short)0, acc[nt], false, false);
    }
  }

#pragma unroll
  for (int nt = 0; nt < NT; ++nt) {
    const int col = (ntBase + nt) * 16 + l16;
    if (col < N) {
      const float bv = bias ? bias[col] : 0.0f;
      float* crow = C + (size_t)(rowBase + half * 8) * N + col;
#pragma unroll
      for (int v = 0; v < 8; ++v) {
        float val = acc[nt][v] + bv;
        if (RELU) val = fmaxf(val, 0.0f);
        crow[(size_t)v * N] = val;
      }
    }
  }
}

// ---------------------------------------------------------------------------
// Fused edge MLP: out = relu(concat(f[r], f[c]) @ Wp1 + bp1) @ Wp2 + bp2.
// One wave per 16 edges; the 16x256 hidden tile lives in LDS as bf16 so the
// 800 MB intermediate never touches HBM.
// ---------------------------------------------------------------------------
__global__ void __launch_bounds__(128)
edge_mlp_kernel(const float* __restrict__ f, const long long* __restrict__ rows,
                const long long* __restrict__ cols,
                const __bf16* __restrict__ Wp1p, const float* __restrict__ bp1,
                const __bf16* __restrict__ Wp2p, const float* __restrict__ bp2,
                float* __restrict__ out, int E) {
  __shared__ __bf16 hid[4][16][256] __attribute__((aligned(16)));

  const int lane = threadIdx.x & 31;
  const int wv   = threadIdx.x >> 5;
  const int half = lane >> 4;
  const int l16  = lane & 15;
  const int edgeBase = (blockIdx.x * 4 + wv) * 16;
  if (edgeBase >= E) return;  // wave-uniform

  const long long rr = rows[edgeBase + l16];
  const long long cc = cols[edgeBase + l16];
  const float* fr = f + (size_t)rr * 128;
  const float* fc = f + (size_t)cc * 128;

  // ---- stage 1: hidden = relu(ef @ Wp1 + bp1), two passes of 128 cols ----
  for (int nblk = 0; nblk < 2; ++nblk) {
    floatx8 acc[8];
#pragma unroll
    for (int nt = 0; nt < 8; ++nt) acc[nt] = 0.0f;

    for (int k0 = 0; k0 < 256; k0 += 32) {
      const int ka = k0 + half * 8;
      const int kb = k0 + 16 + half * 8;
      const float* pa = (ka < 128) ? (fr + ka) : (fc + (ka - 128));
      const float* pb = (kb < 128) ? (fr + kb) : (fc + (kb - 128));
      const bf16x16 a = cvt_frag(pa, pb);
      const __bf16* bt = Wp1p + ((size_t)(k0 >> 5) * 16 + nblk * 8) * 512 + lane * 16;
#pragma unroll
      for (int nt = 0; nt < 8; ++nt) {
        const bf16x16 b = *(const bf16x16*)(bt + (size_t)nt * 512);
        acc[nt] = __builtin_amdgcn_wmma_f32_16x16x32_bf16(false, a, false, b,
                                                          (short)0, acc[nt], false, false);
      }
    }
#pragma unroll
    for (int nt = 0; nt < 8; ++nt) {
      const int col = nblk * 128 + nt * 16 + l16;
      const float bv = bp1[col];
#pragma unroll
      for (int v = 0; v < 8; ++v) {
        hid[wv][half * 8 + v][col] = (__bf16)fmaxf(acc[nt][v] + bv, 0.0f);
      }
    }
  }

  // ---- stage 2: out = hidden @ Wp2 + bp2 (same-wave LDS, hw-ordered) ----
  floatx8 acc2[8];
#pragma unroll
  for (int nt = 0; nt < 8; ++nt) acc2[nt] = 0.0f;

  const __bf16* hrow = &hid[wv][l16][0];
  for (int k0 = 0; k0 < 256; k0 += 32) {
    bf16x8 h0 = *(const bf16x8*)(hrow + k0 + half * 8);
    bf16x8 h1 = *(const bf16x8*)(hrow + k0 + 16 + half * 8);
    bf16x16 a;
#pragma unroll
    for (int i = 0; i < 8; ++i) { a[i] = h0[i]; a[8 + i] = h1[i]; }
    const __bf16* bt = Wp2p + (size_t)(k0 >> 5) * 8 * 512 + lane * 16;
#pragma unroll
    for (int nt = 0; nt < 8; ++nt) {
      const bf16x16 b = *(const bf16x16*)(bt + (size_t)nt * 512);
      acc2[nt] = __builtin_amdgcn_wmma_f32_16x16x32_bf16(false, a, false, b,
                                                         (short)0, acc2[nt], false, false);
    }
  }
#pragma unroll
  for (int nt = 0; nt < 8; ++nt) {
    const int col = nt * 16 + l16;
    const float bv = bp2[col];
    float* orow = out + (size_t)(edgeBase + half * 8) * 128 + col;
#pragma unroll
    for (int v = 0; v < 8; ++v) orow[(size_t)v * 128] = acc2[nt][v] + bv;
  }
}

// --------------------------- GCN normalization -----------------------------
__global__ void deg_init_kernel(float* deg, int n) {
  int t = blockIdx.x * blockDim.x + threadIdx.x;
  if (t < n) deg[t] = 1.0f;  // self-loop contribution
}
__global__ void deg_acc_kernel(const long long* __restrict__ cols, float* deg, int e) {
  int t = blockIdx.x * blockDim.x + threadIdx.x;
  if (t < e)
    __hip_atomic_fetch_add(&deg[(int)cols[t]], 1.0f, __ATOMIC_RELAXED,
                           __HIP_MEMORY_SCOPE_AGENT);
}
__global__ void deg_fin_kernel(float* deg, int n) {
  int t = blockIdx.x * blockDim.x + threadIdx.x;
  if (t < n) deg[t] = rsqrtf(deg[t]);  // deg >= 1 always
}

// --------------------------- scatter (segment sum) -------------------------
__global__ void scatter_self_kernel(const float* __restrict__ xw,
                                    const float* __restrict__ dinv,
                                    float* __restrict__ agg, int total, int fshift) {
  int t = blockIdx.x * blockDim.x + threadIdx.x;
  if (t >= total) return;
  int i = t >> fshift;
  float dv = dinv[i];
  agg[t] = dv * dv * xw[t];
}
__global__ void scatter_edge_kernel(const float* __restrict__ xw,
                                    const long long* __restrict__ rows,
                                    const long long* __restrict__ cols,
                                    const float* __restrict__ dinv,
                                    float* __restrict__ agg,
                                    int total, int fshift, int fmask) {
  int t = blockIdx.x * blockDim.x + threadIdx.x;
  if (t >= total) return;
  int e = t >> fshift;
  int d = t & fmask;
  int rr = (int)rows[e];
  int cc = (int)cols[e];
  float v = dinv[rr] * dinv[cc] * xw[((size_t)rr << fshift) + d];
  __hip_atomic_fetch_add(&agg[((size_t)cc << fshift) + d], v, __ATOMIC_RELAXED,
                         __HIP_MEMORY_SCOPE_AGENT);
}
__global__ void bias_act_kernel(const float* __restrict__ in,
                                const float* __restrict__ bias,
                                float* __restrict__ out, int total, int fmask, int relu) {
  int t = blockIdx.x * blockDim.x + threadIdx.x;
  if (t >= total) return;
  float v = in[t] + bias[t & fmask];
  if (relu) v = fmaxf(v, 0.0f);
  out[t] = v;
}
__global__ void idx_copy_kernel(const long long* __restrict__ idx,
                                float* __restrict__ out, int n) {
  int t = blockIdx.x * blockDim.x + threadIdx.x;
  if (t < n) out[t] = (float)idx[t];
}

// ---------------------------------------------------------------------------
extern "C" void kernel_launch(void* const* d_in, const int* in_sizes, int n_in,
                              void* d_out, int out_size, void* d_ws, size_t ws_size,
                              hipStream_t stream) {
  (void)in_sizes; (void)n_in; (void)out_size; (void)ws_size;
  const int N = 100000, E = 800000;

  const float*     x    = (const float*)d_in[0];
  const long long* eidx = (const long long*)d_in[1];
  const long long* rows = eidx;        // edge_index[0] (source)
  const long long* cols = eidx + E;    // edge_index[1] (target)
  const float* W1  = (const float*)d_in[2];
  const float* b1  = (const float*)d_in[3];
  const float* W2  = (const float*)d_in[4];
  const float* b2  = (const float*)d_in[5];
  const float* Wp1 = (const float*)d_in[6];
  const float* bp1 = (const float*)d_in[7];
  const float* Wp2 = (const float*)d_in[8];
  const float* bp2 = (const float*)d_in[9];
  const float* Wc  = (const float*)d_in[10];
  const float* bc  = (const float*)d_in[11];

  // workspace layout (all fragment regions 32B-aligned)
  float* ws   = (float*)d_ws;
  float* dinv = ws;                       // 100000 (reserve 102400)
  float* bufA = ws + 102400;              // 25.6M floats (xw / hw)
  float* bufB = bufA + 25600000;          // 25.6M floats (agg / h)
  __bf16* W1p  = (__bf16*)(bufB + 25600000);  // 8*16*512
  __bf16* W2p  = W1p  + 8 * 16 * 512;         // 8*8*512
  __bf16* Wp1p = W2p  + 8 * 8 * 512;          // 8*16*512
  __bf16* Wp2p = Wp1p + 8 * 16 * 512;         // 8*8*512
  __bf16* Wcp  = Wp2p + 8 * 8 * 512;          // 4*3*512 (N=40 zero-padded to 48)

  // output layout: f | edge_feats | logits | edge_index
  float* f_out  = (float*)d_out;                 // N*128
  float* ef_out = f_out  + (size_t)N * 128;      // E*128
  float* lg_out = ef_out + (size_t)E * 128;      // N*40
  float* ei_out = lg_out + (size_t)N * 40;       // 2*E

  auto nb = [](long long n, int t) { return (unsigned)((n + t - 1) / t); };

  // ---- pack weights into WMMA B-fragment layout (bf16) ----
  pack_b_kernel<<<nb(8 * 16 * 512, 256), 256, 0, stream>>>(W1,  W1p,  256, 256, 16);
  pack_b_kernel<<<nb(8 * 8 * 512, 256), 256, 0, stream>>>(W2,  W2p,  256, 128, 8);
  pack_b_kernel<<<nb(8 * 16 * 512, 256), 256, 0, stream>>>(Wp1, Wp1p, 256, 256, 16);
  pack_b_kernel<<<nb(8 * 8 * 512, 256), 256, 0, stream>>>(Wp2, Wp2p, 256, 128, 8);
  pack_b_kernel<<<nb(4 * 3 * 512, 256), 256, 0, stream>>>(Wc,  Wcp,  128, 40, 3);

  // ---- symmetric GCN normalization: dinv = rsqrt(1 + indeg) ----
  deg_init_kernel<<<nb(N, 256), 256, 0, stream>>>(dinv, N);
  deg_acc_kernel<<<nb(E, 256), 256, 0, stream>>>(cols, dinv, E);
  deg_fin_kernel<<<nb(N, 256), 256, 0, stream>>>(dinv, N);

  const unsigned gemmRowBlocks = nb(N, 64);  // 64 rows per block

  // ---- conv1: h = relu(segsum(norm * (x@W1)) + b1) ----
  gemm_bf16_wmma<8, false><<<gemmRowBlocks, 128, 0, stream>>>(x, W1p, nullptr, bufA, N, 256, 256, 0, 16);
  gemm_bf16_wmma<8, false><<<gemmRowBlocks, 128, 0, stream>>>(x, W1p, nullptr, bufA, N, 256, 256, 8, 16);
  scatter_self_kernel<<<nb((long long)N * 256, 256), 256, 0, stream>>>(bufA, dinv, bufB, N * 256, 8);
  scatter_edge_kernel<<<nb((long long)E * 256, 256), 256, 0, stream>>>(bufA, rows, cols, dinv, bufB, E * 256, 8, 255);
  bias_act_kernel<<<nb((long long)N * 256, 256), 256, 0, stream>>>(bufB, b1, bufB, N * 256, 255, 1);

  // ---- conv2: f = segsum(norm * (h@W2)) + b2 ----
  gemm_bf16_wmma<8, false><<<gemmRowBlocks, 128, 0, stream>>>(bufB, W2p, nullptr, bufA, N, 256, 128, 0, 8);
  scatter_self_kernel<<<nb((long long)N * 128, 256), 256, 0, stream>>>(bufA, dinv, bufB, N * 128, 7);
  scatter_edge_kernel<<<nb((long long)E * 128, 256), 256, 0, stream>>>(bufA, rows, cols, dinv, bufB, E * 128, 7, 127);
  bias_act_kernel<<<nb((long long)N * 128, 256), 256, 0, stream>>>(bufB, b2, f_out, N * 128, 127, 0);

  // ---- fused edge MLP ----
  edge_mlp_kernel<<<nb(E, 64), 128, 0, stream>>>(f_out, rows, cols, Wp1p, bp1, Wp2p, bp2, ef_out, E);

  // ---- classifier logits = f@Wc + bc ----
  gemm_bf16_wmma<3, false><<<gemmRowBlocks, 128, 0, stream>>>(f_out, Wcp, bc, lg_out, N, 128, 40, 0, 3);

  // ---- edge_index passthrough ----
  idx_copy_kernel<<<nb(2LL * E, 256), 256, 0, stream>>>(eidx, ei_out, 2 * E);
}
